// PC_graph_zwol_PYG_22943715295622
// MI455X (gfx1250) — compile-verified, matrix operand order checked by reading, not run on hardware
//
#include <hip/hip_runtime.h>

// ---------------- problem constants (from reference) ----------------
#define N_V   5000
#define N_E   160000
#define F_W   256
#define T_STEPS 8
#define LR_XC 0.1f
#define N_INT 2500
#define NP    5120          // padded vertex count: multiple of 128 (GEMM M-block)
#define NK    (NP / 32)     // K-steps per GEMM (even, >= 4)

// ---------------- WMMA fragment types ----------------
typedef __attribute__((ext_vector_type(16))) __bf16 v16bf;
typedef __attribute__((ext_vector_type(8)))  float  v8f;

// ---------------- bf16 helpers (RTNE, header-independent) ----------------
__device__ __forceinline__ unsigned short f32_to_bf16(float f) {
    union { float f; unsigned int u; } v; v.f = f;
    unsigned int u = v.u;
    u += 0x7FFFu + ((u >> 16) & 1u);     // round to nearest even
    return (unsigned short)(u >> 16);
}
__device__ __forceinline__ float bf16_to_f32(unsigned short h) {
    union { unsigned int u; float f; } v; v.u = ((unsigned int)h) << 16;
    return v.f;
}

// ---------------- CDNA5 async global->LDS copy (ASYNCcnt-tracked) ----------
__device__ __forceinline__ void async_g2lds_b128(unsigned lds_off, const void* gaddr) {
    asm volatile("global_load_async_to_lds_b128 %0, %1, off"
                 :: "v"(lds_off), "v"(gaddr) : "memory");
}
__device__ __forceinline__ void wait_async_le1() {
    asm volatile("s_wait_asynccnt 0x1" ::: "memory");
}
__device__ __forceinline__ void wait_async_le0() {
    asm volatile("s_wait_asynccnt 0x0" ::: "memory");
}
// Fence-less workgroup barrier: execution sync only. All LDS/VMEM ordering in
// k_gemm is already enforced by s_wait_asynccnt and register-dependency waits,
// so we avoid __syncthreads()'s release fence (s_wait_loadcnt 0x0) which would
// flush the A-fragment prefetch pipeline at every K-step.
__device__ __forceinline__ void wg_barrier() {
    asm volatile("s_barrier_signal -1\n\ts_barrier_wait -1" ::: "memory");
}

// ---------------- zero fill ----------------
__global__ void k_zero_u32(unsigned int* __restrict__ p, size_t n) {
    size_t i = (size_t)blockIdx.x * blockDim.x + threadIdx.x;
    if (i < n) p[i] = 0u;
}

// ---------------- densify: W_eff[d, s] += w[s, d] per edge (bf16, CAS add) ----
__global__ void k_build_w(const float* __restrict__ w, const int* __restrict__ ei,
                          unsigned int* __restrict__ Wb /* bf16 pairs */) {
    int e = blockIdx.x * blockDim.x + threadIdx.x;
    if (e >= N_E) return;
    int s = ei[e];           // edge_index[0][e]  (source)
    int d = ei[N_E + e];     // edge_index[1][e]  (dest)
    float val = w[(size_t)s * N_V + d];
    size_t idx = (size_t)d * NP + s;              // W_eff row = dst, col = src
    unsigned int* word = &Wb[idx >> 1];
    unsigned int shift = (idx & 1) ? 16u : 0u;
    unsigned int old = *word, assumed;
    do {
        assumed = old;
        float cur = bf16_to_f32((unsigned short)((assumed >> shift) & 0xFFFFu));
        unsigned short nb = f32_to_bf16(cur + val);
        unsigned int neww = (assumed & ~(0xFFFFu << shift)) | ((unsigned int)nb << shift);
        old = atomicCAS(word, assumed, neww);
    } while (old != assumed);
}

// ---------------- activation, transposed bf16 store: out[f*NP + s] ------------
// mode 0: tanh(x)      (for prediction pass)
// mode 1: 1 - tanh(x)^2 (for gradient pass)
__global__ void k_act_T(const float* __restrict__ in, unsigned short* __restrict__ out,
                        int mode) {
    int idx = blockIdx.x * blockDim.x + threadIdx.x;
    if (idx >= N_V * F_W) return;
    int f = idx / N_V;
    int s = idx - f * N_V;                 // consecutive threads -> consecutive s: coalesced store
    float t = tanhf(in[(size_t)s * F_W + f]);
    float r = mode ? (1.0f - t * t) : t;
    out[(size_t)f * NP + s] = f32_to_bf16(r);
}

// ---------------- err = vals - pred ----------------
__global__ void k_err(const float* __restrict__ vals, const float* __restrict__ pred,
                      float* __restrict__ err) {
    int idx = blockIdx.x * blockDim.x + threadIdx.x;
    if (idx < N_V * F_W) err[idx] = vals[idx] - pred[idx];
}

// ---------------- vals[0:2500] -= lr * (err - S) ----------------
__global__ void k_update(float* __restrict__ vals, const float* __restrict__ err,
                         const float* __restrict__ S) {
    int idx = blockIdx.x * blockDim.x + threadIdx.x;
    if (idx < N_INT * F_W) vals[idx] -= LR_XC * (err[idx] - S[idx]);
}

// ---------------- bf16 GEMM: C[NxF] = A[NPxNP] * B,  B given as Bt[F x NP] ----
// block = 256 threads (8 waves). Block computes 128 rows x 64 cols.
// Wave w: rows m0 = by*128 + w*16; all waves share f0 = bx*64.
// - B tile (64 f-rows x 32 k) double-buffered in LDS, filled one K-step ahead
//   by global_load_async_to_lds_b128 (ASYNCcnt DMA), drained with
//   s_wait_asynccnt + fence-less barriers.
// - A fragments ping-pong in registers (manual 2x unroll -> no copies),
//   refilled at distance 2 so s_wait_loadcnt covers ~2 K-steps of L2 latency.
// - Last two K-steps peeled: steady-state loop has zero branches.
__global__ __launch_bounds__(256)
void k_gemm(const unsigned short* __restrict__ A,   // W_eff bf16 [NP x NP]
            const unsigned short* __restrict__ Bt,  // act^T  bf16 [F  x NP]
            float* __restrict__ C)                  // f32 [N_V x F]
{
    __shared__ alignas(16) unsigned short lb[2][64 * 40];
    const int tid  = threadIdx.x;
    const int wave = tid >> 5;
    const int lane = tid & 31;
    const int ln   = lane & 15;       // matrix row/col index within fragment
    const int h    = lane >> 4;       // lane-half selects K sub-ranges
    const int f0   = blockIdx.x * 64;
    const int m0   = blockIdx.y * 128 + wave * 16;

    // staging role of this thread: one b128 (16 B) of the 64x32 bf16 tile
    const int srow = tid >> 2;        // 0..63  (f-row within tile)
    const int sch  = tid & 3;         // 0..3   (16-byte chunk within 64-byte row)
    const unsigned short* sgbase = Bt + (size_t)(f0 + srow) * NP + sch * 8;
    const unsigned ldsA = (unsigned)(size_t)&lb[0][srow * 40 + sch * 8];
    const unsigned ldsB = (unsigned)(size_t)&lb[1][srow * 40 + sch * 8];

    union Frag { uint4 q[2]; v16bf v; };
    // per-step A chunk addresses: abase + i*32 (+16) covers K sub-ranges for lane-half h
    const unsigned short* abase = A + (size_t)(m0 + ln) * NP + 8 * h;

    v8f acc[4] = {v8f{}, v8f{}, v8f{}, v8f{}};
    Frag fA[2];

    // prologue: stage B tile 0; preload A fragments for K-steps 0 and 1
    async_g2lds_b128(ldsA, (const void*)sgbase);
    fA[0].q[0] = *(const uint4*)(abase);
    fA[0].q[1] = *(const uint4*)(abase + 16);
    fA[1].q[0] = *(const uint4*)(abase + 32);
    fA[1].q[1] = *(const uint4*)(abase + 48);

    // doStage / doRefill are compile-time constants at every call site, so the
    // steady-state loop body contains no conditional branches.
    auto do_step = [&](int i, int p, bool doStage, bool doRefill) {
        wg_barrier();                          // all reads of lb[p^1] have drained
        if (doStage) {
            // stage next B tile (overlaps this step's WMMAs)
            async_g2lds_b128(p ? ldsA : ldsB,
                             (const void*)(sgbase + (size_t)(i + 1) * 32));
            wait_async_le1();                  // tile i landed (in-order completion)
        } else {
            wait_async_le0();                  // final tile landed
        }
        wg_barrier();                          // tile i visible to all waves

        #pragma unroll
        for (int j = 0; j < 4; ++j) {
            union { uint4 q[2]; v16bf v; } bfr;
            const unsigned short* brow = &lb[p][(16 * j + ln) * 40 + 8 * h];
            bfr.q[0] = *(const uint4*)(brow);
            bfr.q[1] = *(const uint4*)(brow + 16);
            acc[j] = __builtin_amdgcn_wmma_f32_16x16x32_bf16(
                         false, fA[p].v, false, bfr.v, (short)0, acc[j], false, false);
        }
        if (doRefill) {                        // refill fA[p] for K-step i+2
            fA[p].q[0] = *(const uint4*)(abase + (size_t)(i + 2) * 32);
            fA[p].q[1] = *(const uint4*)(abase + (size_t)(i + 2) * 32 + 16);
        }
    };

    #pragma unroll 1
    for (int i = 0; i < NK - 2; i += 2) {      // branch-free steady state
        do_step(i,     0, true, true);
        do_step(i + 1, 1, true, true);
    }
    do_step(NK - 2, 0, true,  false);          // peeled epilogue
    do_step(NK - 1, 1, false, false);

    // store: VGPR r holds row m0 + 8h + r, col f0 + 16j + ln  (ISA C/D layout)
    #pragma unroll
    for (int j = 0; j < 4; ++j) {
        #pragma unroll
        for (int r = 0; r < 8; ++r) {
            int row = m0 + h * 8 + r;
            if (row < N_V) C[(size_t)row * F_W + f0 + 16 * j + ln] = acc[j][r];
        }
    }
}

// ---------------- host-side orchestration ----------------
extern "C" void kernel_launch(void* const* d_in, const int* in_sizes, int n_in,
                              void* d_out, int out_size, void* d_ws, size_t ws_size,
                              hipStream_t stream) {
    const float* values = (const float*)d_in[0];   // [N_V, F]
    const float* w      = (const float*)d_in[1];   // [N_V, N_V]
    const int*   ei     = (const int*)d_in[2];     // [2, E]
    (void)in_sizes; (void)n_in; (void)out_size; (void)ws_size;

    // workspace layout (~67 MB)
    float* vals = (float*)d_ws;                          // N_V*F f32
    float* pred = vals + (size_t)N_V * F_W;              // N_V*F f32 (also S)
    float* errb = pred + (size_t)N_V * F_W;              // N_V*F f32
    unsigned short* Wb = (unsigned short*)(errb + (size_t)N_V * F_W); // NP*NP bf16
    unsigned short* Xt = Wb + (size_t)NP * NP;           // F*NP bf16 (transposed acts)

    const int TB = 256;
    const size_t wWords = (size_t)NP * NP / 2;           // bf16 pairs
    const size_t xWords = (size_t)F_W * NP / 2;
    const int nNF = N_V * F_W;

    // zero W_eff and Xt (pad regions must be 0)
    k_zero_u32<<<dim3((unsigned)((wWords + TB - 1) / TB)), TB, 0, stream>>>((unsigned int*)Wb, wWords);
    k_zero_u32<<<dim3((unsigned)((xWords + TB - 1) / TB)), TB, 0, stream>>>((unsigned int*)Xt, xWords);

    // working copy of values
    hipMemcpyAsync(vals, values, sizeof(float) * (size_t)N_V * F_W,
                   hipMemcpyDeviceToDevice, stream);

    // densify masked weights (handles duplicate edges exactly)
    k_build_w<<<dim3((N_E + TB - 1) / TB), TB, 0, stream>>>(w, ei, (unsigned int*)Wb);

    dim3 ggrid(F_W / 64, NP / 128);   // (4, 40)
    for (int t = 0; t < T_STEPS; ++t) {
        // pred = W_eff @ tanh(vals)
        k_act_T<<<dim3((nNF + TB - 1) / TB), TB, 0, stream>>>(vals, Xt, 0);
        k_gemm <<<ggrid, TB, 0, stream>>>(Wb, Xt, pred);
        // err = vals - pred
        k_err  <<<dim3((nNF + TB - 1) / TB), TB, 0, stream>>>(vals, pred, errb);
        // S = W_eff @ (1 - tanh(err)^2)
        k_act_T<<<dim3((nNF + TB - 1) / TB), TB, 0, stream>>>(errb, Xt, 1);
        k_gemm <<<ggrid, TB, 0, stream>>>(Wb, Xt, pred);
        // vals[0:2500] -= lr * (err - S)
        k_update<<<dim3((N_INT * F_W + TB - 1) / TB), TB, 0, stream>>>(vals, errb, pred);
    }

    hipMemcpyAsync(d_out, vals, sizeof(float) * (size_t)N_V * F_W,
                   hipMemcpyDeviceToDevice, stream);
}
